// EdgeTransition_48533130444941
// MI455X (gfx1250) — compile-verified, matrix operand order checked by reading, not run on hardware
//
#include <hip/hip_runtime.h>
#include <hip/hip_bf16.h>

// ---------------- CDNA5 WMMA types ----------------
typedef __attribute__((ext_vector_type(16))) __bf16 v16bf;
typedef __attribute__((ext_vector_type(8)))  float  v8f;

#define HID   192     // hidden size
#define EOUT  64      // edge-out
#define KT    6       // 192 / 32 k-tiles
#define NT_H  12      // 192 / 16 n-tiles (layers 1,2)
#define NT_O  4       // 64  / 16 n-tiles (final layer)
#define W1_ELEMS  (HID*HID)      // 36864
#define W2_ELEMS  (HID*HID)      // 36864
#define WF_ELEMS  (EOUT*HID)     // 12288
#define W_TOTAL   (W1_ELEMS+W2_ELEMS+WF_ELEMS)   // 86016 bf16 = 172032 B
#define NWAVES 8
#define TPW    8      // tiles per wave -> 64 tiles / 1024 edges per block

// Wave-local LDS fence: per-wave DS ops are tracked by DScnt; waiting for 0
// makes this wave's LDS writes visible to its own lanes (lockstep wave32).
// Much cheaper than a block barrier and keeps the 8 waves decoupled.
__device__ __forceinline__ void lds_fence() {
    asm volatile("s_wait_dscnt 0" ::: "memory");
}

// ------------- kernel 0: convert weights to bf16, transposed [N][K] -------------
__global__ __launch_bounds__(256) void wconv_kernel(
    const float* __restrict__ W1, const float* __restrict__ W2,
    const float* __restrict__ Wf, __bf16* __restrict__ wbf)
{
    int idx = blockIdx.x * 256 + threadIdx.x;
    if (idx < W1_ELEMS) {
        int n = idx / HID, k = idx % HID;
        wbf[n * HID + k] = (__bf16)W1[k * HID + n];
    } else if (idx < W1_ELEMS + W2_ELEMS) {
        int j = idx - W1_ELEMS; int n = j / HID, k = j % HID;
        wbf[W1_ELEMS + n * HID + k] = (__bf16)W2[k * HID + n];
    } else if (idx < W_TOTAL) {
        int j = idx - (W1_ELEMS + W2_ELEMS); int n = j / HID, k = j % HID;
        wbf[W1_ELEMS + W2_ELEMS + n * HID + k] = (__bf16)Wf[k * EOUT + n];
    }
}

// ------------- kernel 1: node projection n = node_embed @ W_init + b_init -> bf16 -------------
__global__ __launch_bounds__(256) void node_proj_kernel(
    const float* __restrict__ node_embed, const float* __restrict__ W_init,
    const float* __restrict__ b_init, __bf16* __restrict__ nbf, int nNodes)
{
    int idx = blockIdx.x * 256 + threadIdx.x;
    int node = idx >> 6;
    int c    = idx & 63;
    if (node >= nNodes) return;
    const float* row = node_embed + (size_t)node * 128;
    float acc = b_init[c];
#pragma unroll 8
    for (int k = 0; k < 128; ++k)
        acc = fmaf(row[k], W_init[k * 64 + c], acc);
    nbf[(size_t)node * 64 + c] = (__bf16)acc;
}

// ------------- fragment loader: row-major [rows][192] bf16, A and B share layout -------------
// 16-bit 16x32 fragment: lane&15 = row (M for A / N for B), K chunks of 8 at
// base = (lane>>4)*8 and base+16  -> two 16-byte LDS loads.
__device__ __forceinline__ v16bf load_frag(const __bf16* buf, int row16base, int kTile, int lane)
{
    int m    = lane & 15;
    int koff = kTile * 32 + ((lane >> 4) << 3);
    const __bf16* p = buf + (size_t)(row16base + m) * HID + koff;
    union { v16bf v; uint4 q[2]; } u;
    u.q[0] = *(const uint4*)(p);
    u.q[1] = *(const uint4*)(p + 16);
    return u.v;
}

// ------------- kernel 2: fused gather + MLP + residual + LayerNorm -------------
__global__ __launch_bounds__(256) void edge_mlp_kernel(
    const float*      __restrict__ edge_embed,
    const long long*  __restrict__ edge_index,   // int64 [2][E]
    const __bf16*     __restrict__ nbf,          // [N][64] bf16
    const __bf16*     __restrict__ wbf,          // W1t | W2t | Wft bf16
    const float* __restrict__ b1, const float* __restrict__ b2,
    const float* __restrict__ bfv,
    const float* __restrict__ gamma, const float* __restrict__ beta,
    float* __restrict__ out, int nEdges)
{
    __shared__ alignas(16) __bf16 sW[W_TOTAL];               // 172032 B
    __shared__ alignas(16) __bf16 sH[NWAVES][16 * HID];      // 6 KB / wave
    __shared__ alignas(16) __bf16 sT[NWAVES][16 * HID];      // 6 KB / wave
    __shared__ float sB1[HID], sB2[HID], sBf[EOUT], sG[EOUT], sBe[EOUT];

    const int tid  = threadIdx.x;
    const int lane = tid & 31;
    const int wave = tid >> 5;

    // ---- stage all weights into LDS (b128 copies), amortized over 64 tiles ----
    {
        const uint4* src = (const uint4*)wbf;
        uint4*       dst = (uint4*)sW;
        const int nq = W_TOTAL * 2 / 16;   // 10752
        for (int i = tid; i < nq; i += 256) dst[i] = src[i];
        for (int i = tid; i < HID; i += 256) { sB1[i] = b1[i]; sB2[i] = b2[i]; }
        for (int i = tid; i < EOUT; i += 256) { sBf[i] = bfv[i]; sG[i] = gamma[i]; sBe[i] = beta[i]; }
    }
    __syncthreads();   // only cross-wave sync needed: weights visible to all waves

    const int numTiles = (nEdges + 15) >> 4;

    __bf16* hbuf = sH[wave];
    __bf16* tbuf = sT[wave];

    const __bf16* sW1 = sW;
    const __bf16* sW2 = sW + W1_ELEMS;
    const __bf16* sWf = sW + W1_ELEMS + W2_ELEMS;

    const int ncolHalf = lane & 15;
    const int mbase    = (lane >> 4) << 3;

    const int tile0 = blockIdx.x * (NWAVES * TPW) + wave;

    for (int it = 0; it < TPW; ++it) {
        const int tile = tile0 + it * NWAVES;
        if (tile >= numTiles) break;
        const int base = tile * 16;

        // ---- gather: h = [edge_embed | n[i] | n[j]] as bf16 [16][192] ----
        {
            int r    = lane >> 1;       // edge row within tile
            int half = lane & 1;        // low/high 32 features
            int e = base + r; if (e >= nEdges) e = nEdges - 1;

            const float4* ef = (const float4*)(edge_embed + (size_t)e * 64 + half * 32);
            __bf16* dE = hbuf + r * HID + half * 32;
#pragma unroll
            for (int q = 0; q < 8; ++q) {
                float4 v = ef[q];
                dE[q * 4 + 0] = (__bf16)v.x;
                dE[q * 4 + 1] = (__bf16)v.y;
                dE[q * 4 + 2] = (__bf16)v.z;
                dE[q * 4 + 3] = (__bf16)v.w;
            }
            long long ni = edge_index[e];
            long long nj = edge_index[(size_t)nEdges + e];
            const uint4* si = (const uint4*)(nbf + (size_t)ni * 64 + half * 32);
            const uint4* sj = (const uint4*)(nbf + (size_t)nj * 64 + half * 32);
            uint4* di = (uint4*)(hbuf + r * HID + 64  + half * 32);
            uint4* dj = (uint4*)(hbuf + r * HID + 128 + half * 32);
            di[0] = si[0]; di[1] = si[1];
            dj[0] = sj[0]; dj[1] = sj[1];
        }
        lds_fence();   // this wave's gather stores visible to all its lanes

        v16bf aF[KT];

        // ---- layer 1: t = relu(h @ W1 + b1) ----
#pragma unroll
        for (int k = 0; k < KT; ++k) aF[k] = load_frag(hbuf, 0, k, lane);
#pragma unroll
        for (int nT = 0; nT < NT_H; ++nT) {
            v8f acc = {};
#pragma unroll
            for (int k = 0; k < KT; ++k) {
                v16bf bF = load_frag(sW1, nT * 16, k, lane);
                acc = __builtin_amdgcn_wmma_f32_16x16x32_bf16(false, aF[k], false, bF,
                                                              (short)0, acc, false, false);
            }
            const int ncol = nT * 16 + ncolHalf;
            const float bias = sB1[ncol];
#pragma unroll
            for (int r = 0; r < 8; ++r) {
                float v = acc[r] + bias;
                v = v > 0.f ? v : 0.f;
                tbuf[(mbase + r) * HID + ncol] = (__bf16)v;
            }
        }
        lds_fence();

        // ---- layer 2: h <- relu(t @ W2 + b2) + h  (residual fused in place) ----
#pragma unroll
        for (int k = 0; k < KT; ++k) aF[k] = load_frag(tbuf, 0, k, lane);
#pragma unroll
        for (int nT = 0; nT < NT_H; ++nT) {
            v8f acc = {};
#pragma unroll
            for (int k = 0; k < KT; ++k) {
                v16bf bF = load_frag(sW2, nT * 16, k, lane);
                acc = __builtin_amdgcn_wmma_f32_16x16x32_bf16(false, aF[k], false, bF,
                                                              (short)0, acc, false, false);
            }
            const int ncol = nT * 16 + ncolHalf;
            const float bias = sB2[ncol];
#pragma unroll
            for (int r = 0; r < 8; ++r) {
                float v = acc[r] + bias;
                v = v > 0.f ? v : 0.f;
                float hres = (float)hbuf[(mbase + r) * HID + ncol];
                hbuf[(mbase + r) * HID + ncol] = (__bf16)(v + hres);
            }
        }
        lds_fence();

        // ---- layer 3: e = (t+h) @ Wf + bf, then LayerNorm over 64 ----
#pragma unroll
        for (int k = 0; k < KT; ++k) aF[k] = load_frag(hbuf, 0, k, lane);
        v8f eF[NT_O];
#pragma unroll
        for (int nT = 0; nT < NT_O; ++nT) {
            v8f acc = {};
#pragma unroll
            for (int k = 0; k < KT; ++k) {
                v16bf bF = load_frag(sWf, nT * 16, k, lane);
                acc = __builtin_amdgcn_wmma_f32_16x16x32_bf16(false, aF[k], false, bF,
                                                              (short)0, acc, false, false);
            }
            const float bias = sBf[nT * 16 + ncolHalf];
#pragma unroll
            for (int r = 0; r < 8; ++r) eF[nT][r] = acc[r] + bias;
        }

        // LayerNorm: each C-fragment row M = mbase + r lives across 16 lanes (4 cols each)
#pragma unroll
        for (int r = 0; r < 8; ++r) {
            float s = 0.f, s2 = 0.f;
#pragma unroll
            for (int nT = 0; nT < NT_O; ++nT) { float x = eF[nT][r]; s += x; s2 += x * x; }
            for (int m = 1; m < 16; m <<= 1) {
                s  += __shfl_xor(s,  m, 16);
                s2 += __shfl_xor(s2, m, 16);
            }
            const float mu  = s * (1.f / 64.f);
            const float var = s2 * (1.f / 64.f) - mu * mu;
            const float rs  = rsqrtf(var + 1e-5f);
            const int erow  = base + mbase + r;
            if (erow < nEdges) {
#pragma unroll
                for (int nT = 0; nT < NT_O; ++nT) {
                    int ncol = nT * 16 + ncolHalf;
                    float x = (eF[nT][r] - mu) * rs * sG[ncol] + sBe[ncol];
                    out[(size_t)erow * 64 + ncol] = x;
                }
            }
        }
        // Per-wave DS ops are in-order: next iteration's gather stores cannot
        // pass this iteration's fragment loads, so no extra fence needed here.
    }
}

// ---------------- host-side launcher ----------------
extern "C" void kernel_launch(void* const* d_in, const int* in_sizes, int n_in,
                              void* d_out, int out_size, void* d_ws, size_t ws_size,
                              hipStream_t stream)
{
    const float*     node_embed = (const float*)d_in[0];
    const float*     edge_embed = (const float*)d_in[1];
    const long long* edge_index = (const long long*)d_in[2];
    const float*     W_init     = (const float*)d_in[3];
    const float*     b_init     = (const float*)d_in[4];
    const float*     W1         = (const float*)d_in[5];
    const float*     b1         = (const float*)d_in[6];
    const float*     W2         = (const float*)d_in[7];
    const float*     b2         = (const float*)d_in[8];
    const float*     Wf         = (const float*)d_in[9];
    const float*     bfv        = (const float*)d_in[10];
    const float*     gamma      = (const float*)d_in[11];
    const float*     beta       = (const float*)d_in[12];
    float*           out        = (float*)d_out;

    const int nNodes = in_sizes[0] / 128;
    const int nEdges = in_sizes[1] / 64;

    // workspace layout: [ nbf : nNodes*64 bf16 ][ wbf : 86016 bf16 ]
    __bf16* nbf = (__bf16*)d_ws;
    __bf16* wbf = (__bf16*)((char*)d_ws + (size_t)nNodes * 64 * sizeof(__bf16));

    // 0) weight convert/transpose
    wconv_kernel<<<(W_TOTAL + 255) / 256, 256, 0, stream>>>(W1, W2, Wf, wbf);

    // 1) node projection -> bf16
    {
        int total = nNodes * 64;
        node_proj_kernel<<<(total + 255) / 256, 256, 0, stream>>>(
            node_embed, W_init, b_init, nbf, nNodes);
    }

    // 2) fused edge MLP: 64 tiles (1024 edges) per block amortize the 172 KB
    //    LDS weight staging; waves run decoupled (no block barriers in loop).
    {
        int numTiles = (nEdges + 15) / 16;
        int blocks   = (numTiles + NWAVES * TPW - 1) / (NWAVES * TPW);
        edge_mlp_kernel<<<blocks, 32 * NWAVES, 0, stream>>>(
            edge_embed, edge_index, nbf, wbf,
            b1, b2, bfv, gamma, beta, out, nEdges);
    }
}
